// GrainNN2_76931454206561
// MI455X (gfx1250) — matrix operand order repaired and checked by reading, not compiled, not run on hardware
//
#include <hip/hip_runtime.h>
#include <hip/hip_bf16.h>
#include <math.h>

namespace {
constexpr int NG   = 30000;
constexpr int NJ   = 60000;
constexpr int EE   = 200000;
constexpr int HD   = 128;
constexpr int FG   = 8;
constexpr int FJ   = 6;
constexpr int NC   = 512;      // 4 gates * H fused columns, order [i,f,c,o]
constexpr int PADN = 262144;   // 2^18 >= EE, bitonic pad
constexpr int KTC  = HD / 32;  // K tiles per GEMM (=4)
constexpr int NGH  = NG * HD;
constexpr int NJH  = NJ * HD;
}

typedef __attribute__((ext_vector_type(16))) __bf16 v16bf;
typedef __attribute__((ext_vector_type(8)))  float  v8f;

__device__ __forceinline__ unsigned short f2bf(float f) {
  unsigned int u = __float_as_uint(f);
  return (unsigned short)((u + 0x7FFFu + ((u >> 16) & 1u)) >> 16);  // RNE truncate
}
__device__ __forceinline__ float sigmoidf(float x) { return 1.f / (1.f + __expf(-x)); }

// ---------------------------------------------------------------------------
// Weight fusion / packing
// ---------------------------------------------------------------------------
// Pack 4 gate weights (each HD x HD fp32, gate order i,f,c,o across columns)
// into the bf16 B-fragment layout for v_wmma_f32_16x16x32_bf16:
// element index e = (((ntile*KT + ktile)*32 + lane)*16 + j)
// lane: col = lane&15, khalf = lane>>4 ; K = ktile*32 + khalf*16 + j
__global__ void k_fuse_pack(const float* si, const float* sf, const float* sc, const float* so,
                            const float* ti, const float* tf, const float* tc, const float* to,
                            unsigned short* out) {
  int e = blockIdx.x * blockDim.x + threadIdx.x;
  int total = (NC / 16) * KTC * 32 * 16;
  if (e >= total) return;
  int j    = e & 15;
  int lane = (e >> 4) & 31;
  int rest = e >> 9;
  int t    = rest % KTC;
  int u    = rest / KTC;
  int ncol = u * 16 + (lane & 15);
  int kk   = t * 32 + (lane >> 4) * 16 + j;
  int g    = ncol >> 7;
  int nl   = ncol & 127;
  const float* S[4] = {si, sf, sc, so};
  const float* T[4] = {ti, tf, tc, to};
  float v = S[g][kk * HD + nl];
  if (ti) v += T[g][kk * HD + nl];
  out[e] = f2bf(v);
}

// Fuse 4 gate input weights (F x HD fp32) into F x NC fp32 (for tiny-K layer-0 matmul)
__global__ void k_fuse_f32(const float* si, const float* sf, const float* sc, const float* so,
                           float* out, int F) {
  int e = blockIdx.x * blockDim.x + threadIdx.x;
  if (e >= F * NC) return;
  int k = e / NC, c = e % NC;
  int g = c >> 7, n = c & 127;
  const float* S[4] = {si, sf, sc, so};
  out[e] = S[g][k * HD + n];
}

__global__ void k_bias3(const float* a, const float* b, const float* c, float* dst) {
  int t = threadIdx.x;  // 128
  dst[t] = a[t] + b[t] + (c ? c[t] : 0.f);
}

// ---------------------------------------------------------------------------
// Scatter-mean (segment mean over edges)
// ---------------------------------------------------------------------------
__global__ void k_scatter(const float* hsrc, const int* src, const int* dst,
                          float* acc, float* cnt) {
  int e = blockIdx.x;
  int t = threadIdx.x;  // 128
  int s = src[e], d = dst[e];
  atomicAdd(&acc[(size_t)d * HD + t], hsrc[(size_t)s * HD + t]);
  if (t == 0) atomicAdd(&cnt[d], 1.0f);
}

__global__ void k_mean_bf16(const float* acc, const float* cnt, unsigned short* out, int n) {
  int i = blockIdx.x * blockDim.x + threadIdx.x;
  if (i >= n * HD) return;
  float c = cnt[i >> 7];
  c = c < 1.f ? 1.f : c;
  out[i] = f2bf(acc[i] / c);
}

__global__ void k_tobf16(const float* in, unsigned short* out, int n) {
  int i = blockIdx.x * blockDim.x + threadIdx.x;
  if (i < n) out[i] = f2bf(in[i]);
}

// ---------------------------------------------------------------------------
// Layer-0 pre-activation init: P = bias + x @ Wx (tiny K = 6 or 8, fp32)
// ---------------------------------------------------------------------------
__global__ void k_initP_xw(const float* bias, const float* x, const float* Wx,
                           float* P, int n, int F) {
  int i = blockIdx.x * blockDim.x + threadIdx.x;
  if (i >= n * NC) return;
  int r = i >> 9, c = i & (NC - 1);
  float a = bias[c];
  const float* xr = x + (size_t)r * F;
  for (int k = 0; k < F; k++) a += xr[k] * Wx[k * NC + c];
  P[i] = a;
}

// ---------------------------------------------------------------------------
// Fused multi-operand WMMA GEMM:
//   C[M x NC] = init + sum_p A_p[M x HD](bf16 row-major) * Bp_p[HD x NC](packed bf16)
// init = column-bias broadcast (loadC==0) or existing C contents (loadC==1).
// Accumulator stays in registers across all operands -> single C write pass.
// grid = (M/16, NC/16/4), block = 128 (4 waves, one 16x16 tile each)
// ---------------------------------------------------------------------------
__global__ void __launch_bounds__(128) k_wmma_fused(
    const unsigned short* A0, const unsigned short* B0,
    const unsigned short* A1, const unsigned short* B1,
    const unsigned short* A2, const unsigned short* B2,
    const unsigned short* A3, const unsigned short* B3,
    const float* bias, float* C, int loadC) {
  int lane = threadIdx.x & 31;
  int wave = threadIdx.x >> 5;
  int mt = blockIdx.x;
  int nt = blockIdx.y * 4 + wave;
  int r  = lane & 15;
  int kg = lane >> 4;

  float* Cp = C + ((size_t)(mt * 16 + kg * 8)) * NC + nt * 16 + r;
  v8f acc;
  if (loadC) {
#pragma unroll
    for (int v = 0; v < 8; v++) acc[v] = Cp[(size_t)v * NC];
  } else {
    float bv = bias[nt * 16 + r];
#pragma unroll
    for (int v = 0; v < 8; v++) acc[v] = bv;
  }

  const unsigned short* As[4] = {A0, A1, A2, A3};
  const unsigned short* Bs[4] = {B0, B1, B2, B3};
  for (int p = 0; p < 4; p++) {
    if (!As[p]) break;
    const unsigned short* Arow = As[p] + ((size_t)(mt * 16 + r)) * HD;
    const unsigned short* Bnt  = Bs[p] + (size_t)nt * KTC * 512;  // 32 lanes * 16 elems
#pragma unroll
    for (int t = 0; t < KTC; t++) {
      union { v16bf v; uint4 u[2]; } a, b;
      // A fragment (contiguous K to match packed B): elem j -> K=kg*16+j
      a.u[0] = *(const uint4*)(Arow + t * 32 + kg * 16);
      a.u[1] = *(const uint4*)(Arow + t * 32 + kg * 16 + 8);
      // B fragment: pre-packed contiguous per lane
      const unsigned short* bp = Bnt + t * 512 + lane * 16;
      b.u[0] = *(const uint4*)(bp);
      b.u[1] = *(const uint4*)(bp + 8);
      acc = __builtin_amdgcn_wmma_f32_16x16x32_bf16(false, a.v, false, b.v,
                                                    (short)0, acc, false, false);
    }
  }
#pragma unroll
  for (int v = 0; v < 8; v++) Cp[(size_t)v * NC] = acc[v];
}

// ---------------------------------------------------------------------------
// LSTM gate pointwise: P cols [0:i 128:f 256:c 384:o]
// ---------------------------------------------------------------------------
__global__ void k_gates(const float* P, const float* c_in,
                        const float* wci, const float* wcf, const float* wco,
                        float* h_out, float* c_out, int n) {
  int i = blockIdx.x * blockDim.x + threadIdx.x;
  if (i >= n * HD) return;
  int f = i & 127;
  int r = i >> 7;
  const float* pr = P + (size_t)r * NC;
  float cv = c_in[i];
  float ig = sigmoidf(pr[f] + wci[f] * cv);
  float fg = sigmoidf(pr[128 + f] + wcf[f] * cv);
  float tg = tanhf(pr[256 + f]);
  float cn = fg * cv + ig * tg;
  float og = sigmoidf(pr[384 + f] + wco[f] * cn);
  h_out[i] = og * tanhf(cn);
  c_out[i] = cn;
}

// ---------------------------------------------------------------------------
// Output heads
// ---------------------------------------------------------------------------
__global__ void k_joint_head(const float* h, const float* W, const float* b, float* out) {
  int n = blockIdx.x * blockDim.x + threadIdx.x;
  if (n >= NJ) return;
  const float* hr = h + (size_t)n * HD;
  float a0 = b[0], a1 = b[1];
  for (int k = 0; k < HD; k++) { float hv = hr[k]; a0 += hv * W[k * 2]; a1 += hv * W[k * 2 + 1]; }
  out[n * 2]     = sigmoidf(a0) - 0.5f;
  out[n * 2 + 1] = sigmoidf(a1) - 0.5f;
}

__global__ void k_grain_a(const float* h, const float* W, const float* b, const float* xg,
                          float* area_raw, float* y1_raw, float* sum_abs) {
  int n = blockIdx.x * blockDim.x + threadIdx.x;
  if (n >= NG) return;
  const float* hr = h + (size_t)n * HD;
  float a0 = b[0], a1 = b[1];
  for (int k = 0; k < HD; k++) { float hv = hr[k]; a0 += hv * W[k * 2]; a1 += hv * W[k * 2 + 1]; }
  float ar = fmaxf(a0 + xg[(size_t)n * FG + 3], 0.f);
  area_raw[n] = ar;
  y1_raw[n]   = fmaxf(a1, 0.f);
  atomicAdd(sum_abs, fabsf(ar));
}

__global__ void k_grain_b(const float* area_raw, const float* y1_raw, const float* xg,
                          const float* sum_abs, float* y_grain, float* event) {
  int n = blockIdx.x * blockDim.x + threadIdx.x;
  if (n >= NG) return;
  float denom = fmaxf(sum_abs[0], 1e-12f);
  float a = area_raw[n] / denom;
  y_grain[n * 2]     = a - xg[(size_t)n * FG + 3];
  y_grain[n * 2 + 1] = y1_raw[n];
  event[n] = (a < 1e-6f) ? 1.0f : 0.0f;
}

// ---------------------------------------------------------------------------
// Edge decoder: concat(h[src],h[dst]) @ W1 + b1 -> relu -> @ W2 + b2 -> sigmoid
// one block (128 threads) per edge; W1 (128 KB) is L2-resident
// ---------------------------------------------------------------------------
__global__ void __launch_bounds__(128) k_edge(const float* hj, const int* econ,
                                              const float* W1, const float* b1,
                                              const float* W2, const float* b2,
                                              float* prob, float* key, int* idx) {
  __shared__ float zs[256];
  __shared__ float red[128];
  int e = blockIdx.x;
  int t = threadIdx.x;
  int s = econ[e], d = econ[EE + e];
  zs[t]       = hj[(size_t)s * HD + t];
  zs[128 + t] = hj[(size_t)d * HD + t];
  __syncthreads();
  float a = b1[t];
  for (int k = 0; k < 256; k++) a += zs[k] * W1[k * HD + t];
  a = fmaxf(a, 0.f);
  red[t] = a * W2[t];
  __syncthreads();
  for (int s2 = 64; s2 > 0; s2 >>= 1) {
    if (t < s2) red[t] += red[t + s2];
    __syncthreads();
  }
  if (t == 0) {
    float p = sigmoidf(red[0] + b2[0]);
    prob[e] = p;
    bool m  = (p > 0.5f) && (s < d);
    key[e]  = m ? p : -1.0f;
    idx[e]  = e;
  }
}

__global__ void k_pad(float* key, int* idx) {
  int i = blockIdx.x * blockDim.x + threadIdx.x + EE;
  if (i < PADN) { key[i] = -3.0e38f; idx[i] = 0; }
}

__global__ void k_bitonic(float* key, int* idx, unsigned j, unsigned k) {
  unsigned i = blockIdx.x * blockDim.x + threadIdx.x;
  if (i >= PADN) return;
  unsigned p = i ^ j;
  if (p > i) {
    bool desc = ((i & k) == 0);  // descending overall
    float ki = key[i], kp = key[p];
    bool sw = desc ? (ki < kp) : (ki > kp);
    if (sw) {
      key[i] = kp; key[p] = ki;
      int tt = idx[i]; idx[i] = idx[p]; idx[p] = tt;
    }
  }
}

__global__ void k_gather(const float* key, const int* idx, const int* econ,
                         float* sp, float* ss, float* ds) {
  int i = blockIdx.x * blockDim.x + threadIdx.x;
  if (i >= EE) return;
  sp[i] = key[i];
  int o = idx[i];
  ss[i] = (float)econ[o];
  ds[i] = (float)econ[EE + o];
}

// ---------------------------------------------------------------------------
// Host side
// ---------------------------------------------------------------------------
namespace {
struct CellW {
  unsigned short *pkWlPull, *pkWrPull, *pkWlPush, *pkWlConn, *pkWrJ, *pkWxG, *pkWxJ;
  float *biasG, *biasJ, *WxG0, *WxJ0;
  const float *wciG, *wcfG, *wcoG, *wciJ, *wcfJ, *wcoJ;
};
inline int cdiv(long long a, int b) { return (int)((a + b - 1) / b); }
}  // namespace

extern "C" void kernel_launch(void* const* d_in, const int* in_sizes, int n_in,
                              void* d_out, int out_size, void* d_ws, size_t ws_size,
                              hipStream_t stream) {
  (void)in_sizes; (void)out_size; (void)ws_size;
  if (n_in < 245) return;  // param-leaf layout assumption violated

  const float* xg   = (const float*)d_in[0];
  const float* xj   = (const float*)d_in[1];
  const int*   econ = (const int*)d_in[2];
  const int*   epul = (const int*)d_in[3];
  const int*   epsh = (const int*)d_in[4];

  // params flattened: dec(116) edge(4) enc(116) lin(4), base = 5
  const int PB = 5;
  const int decB = PB, edgeB = PB + 116, encB = PB + 120, linB = PB + 236;
  // gate leaf offsets within layer (sorted c,f,i,o); fused gate order [i,f,c,o]
  const int GOFF[4] = {28, 13, 0, 43};
  auto LF = [&](int i) -> const float* { return (const float*)d_in[i]; };

  // ---- workspace bump allocator ----
  char*  wsb = (char*)d_ws;
  size_t wo  = 0;
  auto alloc = [&](size_t bytes) -> void* {
    void* p = wsb + wo;
    wo += (bytes + 255) & ~(size_t)255;
    return p;
  };
  // state buffers
  float *hA_g = (float*)alloc(NGH * 4), *cA_g = (float*)alloc(NGH * 4);
  float *hB_g = (float*)alloc(NGH * 4), *cB_g = (float*)alloc(NGH * 4);
  float *hC_g = (float*)alloc(NGH * 4), *cC_g = (float*)alloc(NGH * 4);
  float *hA_j = (float*)alloc(NJH * 4), *cA_j = (float*)alloc(NJH * 4);
  float *hB_j = (float*)alloc(NJH * 4), *cB_j = (float*)alloc(NJH * 4);
  float *hC_j = (float*)alloc(NJH * 4), *cC_j = (float*)alloc(NJH * 4);
  // bf16 operands
  unsigned short* hg16  = (unsigned short*)alloc(NGH * 2);
  unsigned short* hj16  = (unsigned short*)alloc(NJH * 2);
  unsigned short* xg16  = (unsigned short*)alloc(NGH * 2);
  unsigned short* xj16  = (unsigned short*)alloc(NJH * 2);
  unsigned short* mpg16 = (unsigned short*)alloc(NGH * 2);
  unsigned short* mpj16 = (unsigned short*)alloc(NJH * 2);
  unsigned short* mcj16 = (unsigned short*)alloc(NJH * 2);
  // pre-activations (mean accumulators alias into these: used strictly before init)
  float* Pg = (float*)alloc((size_t)NG * NC * 4);
  float* Pj = (float*)alloc((size_t)NJ * NC * 4);
  float* acc_mpg = Pg;              // NGH floats
  float* cnt_g   = Pg + NGH;        // NG floats
  float* acc_mpj = Pj;              // NJH
  float* acc_mcj = Pj + NJH;        // NJH
  float* cnt_j1  = Pj + 2 * (size_t)NJH;       // NJ
  float* cnt_j2  = Pj + 2 * (size_t)NJH + NJ;  // NJ
  // fused weights per cell
  CellW cw[4];
  const size_t PKB = (size_t)(NC / 16) * KTC * 32 * 16;  // 65536 elems
  for (int c = 0; c < 4; c++) {
    cw[c].pkWlPull = (unsigned short*)alloc(PKB * 2);
    cw[c].pkWrPull = (unsigned short*)alloc(PKB * 2);
    cw[c].pkWlPush = (unsigned short*)alloc(PKB * 2);
    cw[c].pkWlConn = (unsigned short*)alloc(PKB * 2);
    cw[c].pkWrJ    = (unsigned short*)alloc(PKB * 2);
    cw[c].pkWxG    = (unsigned short*)alloc(PKB * 2);
    cw[c].pkWxJ    = (unsigned short*)alloc(PKB * 2);
    cw[c].biasG    = (float*)alloc(NC * 4);
    cw[c].biasJ    = (float*)alloc(NC * 4);
    cw[c].WxG0     = (float*)alloc((size_t)FG * NC * 4);
    cw[c].WxJ0     = (float*)alloc((size_t)FJ * NC * 4);
  }
  float* area_raw = (float*)alloc(NG * 4);
  float* y1_raw   = (float*)alloc(NG * 4);
  float* sum_abs  = (float*)alloc(4);
  float* skey     = (float*)alloc((size_t)PADN * 4);
  int*   sidx     = (int*)alloc((size_t)PADN * 4);

  // ---- weight prep (cells: 0=enc0, 1=enc1, 2=dec0, 3=dec1) ----
  const int cellBase[4] = {encB, encB + 58, decB, decB + 58};
  const bool cellL0[4]  = {true, false, true, false};
  for (int c = 0; c < 4; c++) {
    int cb = cellBase[c];
    const float *pWl[4], *pWr[4], *uWl[4], *uWr[4], *cWl[4], *cWr[4], *Wg[4], *Wj[4];
    for (int g = 0; g < 4; g++) {
      int b = cb + GOFF[g];
      Wg[g]  = LF(b + 0);  Wj[g]  = LF(b + 1);
      cWl[g] = LF(b + 4);  cWr[g] = LF(b + 5);
      pWl[g] = LF(b + 7);  pWr[g] = LF(b + 8);
      uWl[g] = LF(b + 10); uWr[g] = LF(b + 11);
    }
    dim3 pg(cdiv(PKB, 256)), pb(256);
    k_fuse_pack<<<pg, pb, 0, stream>>>(pWl[0], pWl[1], pWl[2], pWl[3], nullptr, nullptr, nullptr, nullptr, cw[c].pkWlPull);
    k_fuse_pack<<<pg, pb, 0, stream>>>(pWr[0], pWr[1], pWr[2], pWr[3], nullptr, nullptr, nullptr, nullptr, cw[c].pkWrPull);
    k_fuse_pack<<<pg, pb, 0, stream>>>(uWl[0], uWl[1], uWl[2], uWl[3], nullptr, nullptr, nullptr, nullptr, cw[c].pkWlPush);
    k_fuse_pack<<<pg, pb, 0, stream>>>(cWl[0], cWl[1], cWl[2], cWl[3], nullptr, nullptr, nullptr, nullptr, cw[c].pkWlConn);
    k_fuse_pack<<<pg, pb, 0, stream>>>(uWr[0], uWr[1], uWr[2], uWr[3], cWr[0], cWr[1], cWr[2], cWr[3], cw[c].pkWrJ);
    if (cellL0[c]) {
      k_fuse_f32<<<cdiv((size_t)FG * NC, 256), 256, 0, stream>>>(Wg[0], Wg[1], Wg[2], Wg[3], cw[c].WxG0, FG);
      k_fuse_f32<<<cdiv((size_t)FJ * NC, 256), 256, 0, stream>>>(Wj[0], Wj[1], Wj[2], Wj[3], cw[c].WxJ0, FJ);
    } else {
      k_fuse_pack<<<pg, pb, 0, stream>>>(Wg[0], Wg[1], Wg[2], Wg[3], nullptr, nullptr, nullptr, nullptr, cw[c].pkWxG);
      k_fuse_pack<<<pg, pb, 0, stream>>>(Wj[0], Wj[1], Wj[2], Wj[3], nullptr, nullptr, nullptr, nullptr, cw[c].pkWxJ);
    }
    for (int g = 0; g < 4; g++) {
      int b = cb + GOFF[g];
      k_bias3<<<1, 128, 0, stream>>>(LF(b + 9),  LF(b + 2), nullptr,   cw[c].biasG + g * 128);
      k_bias3<<<1, 128, 0, stream>>>(LF(b + 12), LF(b + 6), LF(b + 3), cw[c].biasJ + g * 128);
    }
    cw[c].wciG = LF(cb + 28 + 13); cw[c].wciJ = LF(cb + 28 + 14);
    cw[c].wcfG = LF(cb + 13 + 13); cw[c].wcfJ = LF(cb + 13 + 14);
    cw[c].wcoG = LF(cb + 43 + 13); cw[c].wcoJ = LF(cb + 43 + 14);
  }

  // zero-state buffers (hC,cC serve as the zero state for the encoder)
  hipMemsetAsync(hC_g, 0, (size_t)NGH * 4, stream);
  hipMemsetAsync(cC_g, 0, (size_t)NGH * 4, stream);
  hipMemsetAsync(hC_j, 0, (size_t)NJH * 4, stream);
  hipMemsetAsync(cC_j, 0, (size_t)NJH * 4, stream);

  typedef const unsigned short* U16P;
  auto fused = [&](float* C, int M, const float* bias, int loadC,
                   U16P A0, U16P B0, U16P A1 = nullptr, U16P B1 = nullptr,
                   U16P A2 = nullptr, U16P B2 = nullptr, U16P A3 = nullptr, U16P B3 = nullptr) {
    dim3 grid(M / 16, NC / 64);  // 4 waves/block, one 16x16 tile each
    k_wmma_fused<<<grid, 128, 0, stream>>>(A0, B0, A1, B1, A2, B2, A3, B3, bias, C, loadC);
  };

  auto run_cell = [&](const CellW& w, bool layer0, bool hzero,
                      const float* xgIn, const float* xjIn,
                      const float* hgI, const float* cgI, const float* hjI, const float* cjI,
                      float* hgO, float* cgO, float* hjO, float* cjO) {
    if (!hzero) {
      hipMemsetAsync(acc_mpg, 0, (size_t)NGH * 4, stream);
      hipMemsetAsync(acc_mpj, 0, (size_t)NJH * 4, stream);
      hipMemsetAsync(acc_mcj, 0, (size_t)NJH * 4, stream);
      hipMemsetAsync(cnt_g, 0, NG * 4, stream);
      hipMemsetAsync(cnt_j1, 0, NJ * 4, stream);
      hipMemsetAsync(cnt_j2, 0, NJ * 4, stream);
      k_scatter<<<EE, 128, 0, stream>>>(hjI, epul, epul + EE, acc_mpg, cnt_g);   // pull: joint->grain
      k_scatter<<<EE, 128, 0, stream>>>(hgI, epsh, epsh + EE, acc_mpj, cnt_j1);  // push: grain->joint
      k_scatter<<<EE, 128, 0, stream>>>(hjI, econ, econ + EE, acc_mcj, cnt_j2);  // connect: joint->joint
      k_mean_bf16<<<cdiv(NGH, 256), 256, 0, stream>>>(acc_mpg, cnt_g, mpg16, NG);
      k_mean_bf16<<<cdiv(NJH, 256), 256, 0, stream>>>(acc_mpj, cnt_j1, mpj16, NJ);
      k_mean_bf16<<<cdiv(NJH, 256), 256, 0, stream>>>(acc_mcj, cnt_j2, mcj16, NJ);
      k_tobf16<<<cdiv(NGH, 256), 256, 0, stream>>>(hgI, hg16, NGH);
      k_tobf16<<<cdiv(NJH, 256), 256, 0, stream>>>(hjI, hj16, NJH);
    }
    if (layer0) {
      // P = bias + x@Wx (fp32, tiny K), then conv terms accumulate on top (loadC=1)
      k_initP_xw<<<cdiv((size_t)NG * NC, 256), 256, 0, stream>>>(w.biasG, xgIn, w.WxG0, Pg, NG, FG);
      k_initP_xw<<<cdiv((size_t)NJ * NC, 256), 256, 0, stream>>>(w.biasJ, xjIn, w.WxJ0, Pj, NJ, FJ);
      if (!hzero) {
        fused(Pg, NG, w.biasG, 1, mpg16, w.pkWlPull, hg16, w.pkWrPull);
        fused(Pj, NJ, w.biasJ, 1, mpj16, w.pkWlPush, mcj16, w.pkWlConn, hj16, w.pkWrJ);
      }
    } else {
      k_tobf16<<<cdiv(NGH, 256), 256, 0, stream>>>(xgIn, xg16, NGH);
      k_tobf16<<<cdiv(NJH, 256), 256, 0, stream>>>(xjIn, xj16, NJH);
      if (hzero) {
        fused(Pg, NG, w.biasG, 0, xg16, w.pkWxG);
        fused(Pj, NJ, w.biasJ, 0, xj16, w.pkWxJ);
      } else {
        fused(Pg, NG, w.biasG, 0, xg16, w.pkWxG, mpg16, w.pkWlPull, hg16, w.pkWrPull);
        fused(Pj, NJ, w.biasJ, 0, xj16, w.pkWxJ, mpj16, w.pkWlPush, mcj16, w.pkWlConn, hj16, w.pkWrJ);
      }
    }
    k_gates<<<cdiv(NGH, 256), 256, 0, stream>>>(Pg, cgI, w.wciG, w.wcfG, w.wcoG, hgO, cgO, NG);
    k_gates<<<cdiv(NJH, 256), 256, 0, stream>>>(Pj, cjI, w.wciJ, w.wcfJ, w.wcoJ, hjO, cjO, NJ);
  };

  // enc0: zero state, x = features -> A
  run_cell(cw[0], true, true, xg, xj, hC_g, cC_g, hC_j, cC_j, hA_g, cA_g, hA_j, cA_j);
  // enc1: zero state, x = hA -> B
  run_cell(cw[1], false, true, hA_g, hA_j, hC_g, cC_g, hC_j, cC_j, hB_g, cB_g, hB_j, cB_j);
  // dec0: state = A (enc layer0), x = features -> C
  run_cell(cw[2], true, false, xg, xj, hA_g, cA_g, hA_j, cA_j, hC_g, cC_g, hC_j, cC_j);
  // dec1: state = B (enc layer1), x = hC -> A (final)
  run_cell(cw[3], false, false, hC_g, hC_j, hB_g, cB_g, hB_j, cB_j, hA_g, cA_g, hA_j, cA_j);

  // ---- output heads ----
  float* out  = (float*)d_out;
  float* o_yj = out;                 // NJ*2
  float* o_yg = out + 2 * NJ;        // NG*2
  float* o_ev = out + 2 * NJ + 2 * NG;
  float* o_ep = o_ev + NG;           // EE
  float* o_sp = o_ep + EE;           // EE
  float* o_ss = o_sp + EE;           // EE
  float* o_ds = o_ss + EE;           // EE

  k_joint_head<<<cdiv(NJ, 128), 128, 0, stream>>>(hA_j, LF(linB + 2), LF(linB + 3), o_yj);
  hipMemsetAsync(sum_abs, 0, 4, stream);
  k_grain_a<<<cdiv(NG, 128), 128, 0, stream>>>(hA_g, LF(linB + 0), LF(linB + 1), xg,
                                               area_raw, y1_raw, sum_abs);
  k_grain_b<<<cdiv(NG, 128), 128, 0, stream>>>(area_raw, y1_raw, xg, sum_abs, o_yg, o_ev);

  k_edge<<<EE, 128, 0, stream>>>(hA_j, econ, LF(edgeB + 0), LF(edgeB + 2),
                                 LF(edgeB + 1), LF(edgeB + 3), o_ep, skey, sidx);
  k_pad<<<cdiv(PADN - EE, 256), 256, 0, stream>>>(skey, sidx);
  for (unsigned k = 2; k <= (unsigned)PADN; k <<= 1)
    for (unsigned j = k >> 1; j > 0; j >>= 1)
      k_bitonic<<<PADN / 256, 256, 0, stream>>>(skey, sidx, j, k);
  k_gather<<<cdiv(EE, 256), 256, 0, stream>>>(skey, sidx, econ, o_sp, o_ss, o_ds);
}
